// MTCNN_21586505629984
// MI455X (gfx1250) — compile-verified, matrix operand order checked by reading, not run on hardware
//
#include <hip/hip_runtime.h>
#include <hip/hip_bf16.h>
#include <math.h>

// ---------------------------------------------------------------------------
// MTCNN PNet pyramid for MI455X (gfx1250, wave32).
// conv3 (3x3, 16->32) runs as implicit GEMM on v_wmma_f32_16x16x32_f16:
//   M = 16 output pixels per wave, N = 32 output channels (2 WMMA tiles),
//   K = 3*3*16 = 144 padded to 160 (5 chunks of K=32).
// A-fragment halves are contiguous 16B runs of the f16 NHWC conv2 output, so
// each lane issues two global_load_b128 per K-chunk. 1x1 heads + softmax +
// box/mask generation are fused behind an LDS transpose of the accumulators.
// Workspace requirement: ~120 MB (see kernel_launch partition).
// ---------------------------------------------------------------------------

typedef __attribute__((ext_vector_type(16))) _Float16 v16h;
typedef __attribute__((ext_vector_type(8)))  float    v8f;

__device__ __forceinline__ float lrelu(float x) { return x > 0.f ? x : 0.3f * x; }

// ---------------- antialiased linear resize (JAX semantics), separable -----
__global__ void resize_h_kernel(const float* __restrict__ in, float* __restrict__ out,
                                int hs, int inH, int W, int B) {
    int n = blockIdx.x * blockDim.x + threadIdx.x;
    int total = B * hs * W;
    if (n >= total) return;
    int x = n % W; int t = n / W; int oy = t % hs; int b = t / hs;
    float inv = (float)inH / (float)hs;
    float sample = ((float)oy + 0.5f) * inv - 0.5f;
    float kscale = inv > 1.f ? inv : 1.f;
    int i0 = (int)ceilf(sample - kscale);  if (i0 < 0) i0 = 0;
    int i1 = (int)floorf(sample + kscale); if (i1 > inH - 1) i1 = inH - 1;
    float a0 = 0.f, a1 = 0.f, a2 = 0.f, wsum = 0.f;
    for (int i = i0; i <= i1; ++i) {
        float w = 1.f - fabsf(sample - (float)i) / kscale;
        if (w <= 0.f) continue;
        const float* p = in + (((size_t)b * inH + i) * W + x) * 3;
        a0 += w * p[0]; a1 += w * p[1]; a2 += w * p[2]; wsum += w;
    }
    float r = wsum > 0.f ? 1.f / wsum : 0.f;
    float* q = out + (((size_t)b * hs + oy) * W + x) * 3;
    q[0] = a0 * r; q[1] = a1 * r; q[2] = a2 * r;
}

__global__ void resize_w_kernel(const float* __restrict__ in, float* __restrict__ out,
                                int hs, int inW, int ws, int B) {
    int n = blockIdx.x * blockDim.x + threadIdx.x;
    int total = B * hs * ws;
    if (n >= total) return;
    int ox = n % ws; int t = n / ws; int oy = t % hs; int b = t / hs;
    float inv = (float)inW / (float)ws;
    float sample = ((float)ox + 0.5f) * inv - 0.5f;
    float kscale = inv > 1.f ? inv : 1.f;
    int i0 = (int)ceilf(sample - kscale);  if (i0 < 0) i0 = 0;
    int i1 = (int)floorf(sample + kscale); if (i1 > inW - 1) i1 = inW - 1;
    float a0 = 0.f, a1 = 0.f, a2 = 0.f, wsum = 0.f;
    for (int i = i0; i <= i1; ++i) {
        float w = 1.f - fabsf(sample - (float)i) / kscale;
        if (w <= 0.f) continue;
        const float* p = in + (((size_t)b * hs + oy) * inW + i) * 3;
        a0 += w * p[0]; a1 += w * p[1]; a2 += w * p[2]; wsum += w;
    }
    float r = wsum > 0.f ? 1.f / wsum : 0.f;
    float* q = out + (((size_t)b * hs + oy) * ws + ox) * 3;
    q[0] = (a0 * r - 127.5f) * (1.f / 128.f);
    q[1] = (a1 * r - 127.5f) * (1.f / 128.f);
    q[2] = (a2 * r - 127.5f) * (1.f / 128.f);
}

// ---------------- conv1: 3x3, 3->10, VALID, leaky-relu ---------------------
__global__ void conv1_kernel(const float* __restrict__ img, const float* __restrict__ w1,
                             const float* __restrict__ b1, float* __restrict__ out,
                             int hs, int ws, int c1h, int c1w, int B) {
    __shared__ float sw[270];
    __shared__ float sb[10];
    for (int i = threadIdx.x; i < 270; i += blockDim.x) sw[i] = w1[i];
    for (int i = threadIdx.x; i < 10;  i += blockDim.x) sb[i] = b1[i];
    __syncthreads();
    int n = blockIdx.x * blockDim.x + threadIdx.x;
    int total = B * c1h * c1w;
    if (n >= total) return;
    int x = n % c1w; int t = n / c1w; int y = t % c1h; int b = t / c1h;
    float acc[10];
    #pragma unroll
    for (int co = 0; co < 10; ++co) acc[co] = sb[co];
    for (int ky = 0; ky < 3; ++ky)
        for (int kx = 0; kx < 3; ++kx) {
            const float* p = img + (((size_t)b * hs + y + ky) * ws + x + kx) * 3;
            #pragma unroll
            for (int ci = 0; ci < 3; ++ci) {
                float v = p[ci];
                const float* wrow = sw + ((ky * 3 + kx) * 3 + ci) * 10;
                #pragma unroll
                for (int co = 0; co < 10; ++co) acc[co] += v * wrow[co];
            }
        }
    float* q = out + (size_t)n * 10;
    #pragma unroll
    for (int co = 0; co < 10; ++co) q[co] = lrelu(acc[co]);
}

// ---------------- maxpool 2x2 stride 2, SAME -------------------------------
__global__ void pool_kernel(const float* __restrict__ in, float* __restrict__ out,
                            int c1h, int c1w, int ph, int pw, int B) {
    int n = blockIdx.x * blockDim.x + threadIdx.x;
    int total = B * ph * pw;
    if (n >= total) return;
    int x = n % pw; int t = n / pw; int y = t % ph; int b = t / ph;
    float* q = out + (size_t)n * 10;
    float acc[10];
    #pragma unroll
    for (int c = 0; c < 10; ++c) acc[c] = -3.402823466e+38f;
    for (int dy = 0; dy < 2; ++dy) {
        int iy = 2 * y + dy; if (iy >= c1h) continue;
        for (int dx = 0; dx < 2; ++dx) {
            int ix = 2 * x + dx; if (ix >= c1w) continue;
            const float* p = in + (((size_t)b * c1h + iy) * c1w + ix) * 10;
            #pragma unroll
            for (int c = 0; c < 10; ++c) acc[c] = fmaxf(acc[c], p[c]);
        }
    }
    #pragma unroll
    for (int c = 0; c < 10; ++c) q[c] = acc[c];
}

// ---------------- conv2: 3x3, 10->16, VALID, leaky-relu, f16 out -----------
__global__ void conv2_kernel(const float* __restrict__ in, const float* __restrict__ w2,
                             const float* __restrict__ b2, _Float16* __restrict__ out,
                             int ph, int pw, int c2h, int c2w, int B) {
    __shared__ float sw[1440];
    __shared__ float sb[16];
    for (int i = threadIdx.x; i < 1440; i += blockDim.x) sw[i] = w2[i];
    for (int i = threadIdx.x; i < 16;   i += blockDim.x) sb[i] = b2[i];
    __syncthreads();
    int n = blockIdx.x * blockDim.x + threadIdx.x;
    int total = B * c2h * c2w;
    if (n >= total) return;
    int x = n % c2w; int t = n / c2w; int y = t % c2h; int b = t / c2h;
    float acc[16];
    #pragma unroll
    for (int co = 0; co < 16; ++co) acc[co] = sb[co];
    for (int ky = 0; ky < 3; ++ky)
        for (int kx = 0; kx < 3; ++kx) {
            const float* p = in + (((size_t)b * ph + y + ky) * pw + x + kx) * 10;
            #pragma unroll
            for (int ci = 0; ci < 10; ++ci) {
                float v = p[ci];
                const float* wrow = sw + ((ky * 3 + kx) * 10 + ci) * 16;
                #pragma unroll
                for (int co = 0; co < 16; ++co) acc[co] += v * wrow[co];
            }
        }
    _Float16* q = out + (size_t)n * 16;
    #pragma unroll
    for (int co = 0; co < 16; ++co) q[co] = (_Float16)lrelu(acc[co]);
}

// ---------------- pack w3 (3,3,16,32 HWIO f32) -> WMMA B fragments ---------
// Layout [kc][nhalf][lane][elem], K = kc*32 + kin, where per ISA 16-bit
// operand layout: lane<16 holds K {0..7,16..23}, lane>=16 holds {8..15,24..31}.
__global__ void pack_w3_kernel(const float* __restrict__ w3, _Float16* __restrict__ w3p) {
    for (int idx = blockIdx.x * blockDim.x + threadIdx.x; idx < 5 * 2 * 32 * 16;
         idx += gridDim.x * blockDim.x) {
        int e    = idx & 15;
        int lane = (idx >> 4) & 31;
        int half = (idx >> 9) & 1;
        int kc   = idx >> 10;
        int n     = half * 16 + (lane & 15);
        int khalf = (lane >= 16) ? 8 : 0;
        int kin   = (e < 8) ? (khalf + e) : (16 + khalf + (e - 8));
        int k     = kc * 32 + kin;
        float v = 0.f;
        if (k < 144) {
            int tap = k >> 4;       // 0..8 = ky*3+kx
            int c   = k & 15;       // input channel
            v = w3[((size_t)tap * 16 + c) * 32 + n];
        }
        w3p[idx] = (_Float16)v;
    }
}

// ---------------- conv3 (WMMA) + heads (softmax/reg/boxes/mask), fused -----
__global__ __launch_bounds__(128) void conv3_heads_kernel(
    const _Float16* __restrict__ c2, const _Float16* __restrict__ w3p,
    const float* __restrict__ b3,
    const float* __restrict__ wp, const float* __restrict__ bp,
    const float* __restrict__ wr, const float* __restrict__ br,
    float* __restrict__ boxes, float* __restrict__ regOut, float* __restrict__ maskOut,
    int c2h, int c2w, int c3h, int c3w, int tilesPerRow, int nTiles,
    float scaleF, int scaleOff, int totalAll) {
    __shared__ float tile[4][16][33];
    __shared__ float s_b3[32], s_wp[64], s_bp[2], s_wr[128], s_br[4];
    for (int i = threadIdx.x; i < 32;  i += 128) s_b3[i] = b3[i];
    for (int i = threadIdx.x; i < 64;  i += 128) s_wp[i] = wp[i];
    for (int i = threadIdx.x; i < 2;   i += 128) s_bp[i] = bp[i];
    for (int i = threadIdx.x; i < 128; i += 128) s_wr[i] = wr[i];
    for (int i = threadIdx.x; i < 4;   i += 128) s_br[i] = br[i];
    __syncthreads();

    const int wave = threadIdx.x >> 5;
    const int lane = threadIdx.x & 31;
    int tileId = blockIdx.x * 4 + wave;
    int t = tileId < nTiles ? tileId : 0;       // clamp (predicated stores later)
    int bIdx = t / (c3h * tilesPerRow);
    int rem  = t % (c3h * tilesPerRow);
    int y    = rem / tilesPerRow;
    int x0   = (rem % tilesPerRow) << 4;
    int m    = lane & 15;
    int xm   = x0 + m; if (xm > c3w - 1) xm = c3w - 1;   // overhang clamp
    int chOff = (lane >> 4) << 3;               // 0 or 8: lane's K sub-range

    v8f acc0 = {};
    v8f acc1 = {};
    const size_t rowStride = (size_t)c2w * 16;
    const _Float16* base = c2 + ((size_t)bIdx * c2h + y) * rowStride + (size_t)xm * 16 + chOff;

    #pragma unroll
    for (int kc = 0; kc < 5; ++kc) {
        int t0 = kc * 2, t1 = t0 + 1;           // the two 16-channel taps in this K chunk
        union { v16h h; uint4 q[2]; } a;
        {
            int ky = t0 / 3, kx = t0 - 3 * (t0 / 3);
            a.q[0] = *(const uint4*)(base + (size_t)ky * rowStride + kx * 16);
        }
        if (t1 < 9) {
            int ky = t1 / 3, kx = t1 - 3 * (t1 / 3);
            a.q[1] = *(const uint4*)(base + (size_t)ky * rowStride + kx * 16);
        } else {
            a.q[1] = make_uint4(0u, 0u, 0u, 0u); // K padding (144 -> 160)
        }
        v16h bf0 = *(const v16h*)(w3p + ((size_t)(kc * 2 + 0) * 32 + lane) * 16);
        v16h bf1 = *(const v16h*)(w3p + ((size_t)(kc * 2 + 1) * 32 + lane) * 16);
        acc0 = __builtin_amdgcn_wmma_f32_16x16x32_f16(false, a.h, false, bf0,
                                                      (short)0, acc0, false, false);
        acc1 = __builtin_amdgcn_wmma_f32_16x16x32_f16(false, a.h, false, bf1,
                                                      (short)0, acc1, false, false);
    }

    // D layout: lane<16 -> N=lane, rows M=0..7; lane>=16 -> N=lane-16, rows M=8..15.
    int nb = lane & 15;
    int mb = (lane >> 4) << 3;
    #pragma unroll
    for (int r = 0; r < 8; ++r) {
        tile[wave][mb + r][nb]      = lrelu(acc0[r] + s_b3[nb]);
        tile[wave][mb + r][16 + nb] = lrelu(acc1[r] + s_b3[16 + nb]);
    }
    __syncthreads();

    // heads: one thread per output pixel (4 waves x 16 pixels = 64 pixels/block)
    int tid = threadIdx.x;
    if (tid < 64) {
        int w = tid >> 4, mm = tid & 15;
        int tId = blockIdx.x * 4 + w;
        if (tId < nTiles) {
            int b2 = tId / (c3h * tilesPerRow);
            int rm = tId % (c3h * tilesPerRow);
            int yy = rm / tilesPerRow;
            int xx = (rm % tilesPerRow) * 16 + mm;
            if (xx < c3w) {
                float l0 = s_bp[0], l1 = s_bp[1];
                float r0 = s_br[0], r1 = s_br[1], r2 = s_br[2], r3 = s_br[3];
                #pragma unroll
                for (int c = 0; c < 32; ++c) {
                    float v = tile[w][mm][c];
                    l0 += v * s_wp[c * 2 + 0]; l1 += v * s_wp[c * 2 + 1];
                    r0 += v * s_wr[c * 4 + 0]; r1 += v * s_wr[c * 4 + 1];
                    r2 += v * s_wr[c * 4 + 2]; r3 += v * s_wr[c * 4 + 3];
                }
                float score = 1.f / (1.f + expf(l0 - l1));   // softmax[...,1]
                int i = yy * c3w + xx;
                size_t pos = (size_t)b2 * totalAll + (size_t)scaleOff + i;
                float* bx = boxes + pos * 5;
                bx[0] = rintf(((float)yy * 2.f + 1.f) / scaleF);
                bx[1] = rintf(((float)xx * 2.f + 1.f) / scaleF);
                bx[2] = rintf(((float)yy * 2.f + 12.f) / scaleF);
                bx[3] = rintf(((float)xx * 2.f + 12.f) / scaleF);
                bx[4] = score;
                float* rg = regOut + pos * 4;
                rg[0] = r0; rg[1] = r1; rg[2] = r2; rg[3] = r3;
                maskOut[pos] = score >= 0.6f ? 1.f : 0.f;
            }
        }
    }
}

static inline unsigned cdivu(long long a, long long b) { return (unsigned)((a + b - 1) / b); }

extern "C" void kernel_launch(void* const* d_in, const int* in_sizes, int n_in,
                              void* d_out, int out_size, void* d_ws, size_t ws_size,
                              hipStream_t stream) {
    (void)in_sizes; (void)n_in; (void)out_size; (void)ws_size;
    const float* inp = (const float*)d_in[0];
    const float* w1 = (const float*)d_in[1];  const float* b1 = (const float*)d_in[2];
    const float* w2 = (const float*)d_in[3];  const float* b2 = (const float*)d_in[4];
    const float* w3 = (const float*)d_in[5];  const float* b3 = (const float*)d_in[6];
    const float* wp = (const float*)d_in[7];  const float* bp = (const float*)d_in[8];
    const float* wr = (const float*)d_in[9];  const float* br = (const float*)d_in[10];
    float* out = (float*)d_out;

    const int B = 4, H = 1080, W = 1080;

    // pyramid scales (double math mirrors python)
    double scales[16]; int ns = 0;
    { double sc = 12.0 / 20.0, mm = sc * (double)H;
      while (mm >= 12.0 && ns < 16) { scales[ns++] = sc; sc *= 0.709; mm *= 0.709; } }

    int HS[16], WS[16], C1H[16], C1W[16], PH[16], PW[16], C2H[16], C2W[16],
        C3H[16], C3W[16], OFF[16];
    int totalAll = 0;
    for (int s = 0; s < ns; ++s) {
        HS[s] = (int)ceil((double)H * scales[s]);
        WS[s] = (int)ceil((double)W * scales[s]);
        C1H[s] = HS[s] - 2; C1W[s] = WS[s] - 2;
        PH[s] = (C1H[s] + 1) / 2; PW[s] = (C1W[s] + 1) / 2;
        C2H[s] = PH[s] - 2; C2W[s] = PW[s] - 2;
        C3H[s] = C2H[s] - 2; C3W[s] = C2W[s] - 2;
        OFF[s] = totalAll; totalAll += C3H[s] * C3W[s];
    }

    // workspace partition (sized by largest scale; ~120 MB total)
    char* wsB = (char*)d_ws;
    size_t o = 0;
    auto suballoc = [&](size_t bytes) -> void* {
        void* p = wsB + o; o = (o + bytes + 255) & ~(size_t)255; return p; };
    float*    imgBuf  = (float*)   suballoc((size_t)B * HS[0] * WS[0] * 3 * sizeof(float));
    float*    c1Buf   = (float*)   suballoc((size_t)B * C1H[0] * C1W[0] * 10 * sizeof(float));
    float*    poolBuf = (float*)   suballoc((size_t)B * PH[0] * PW[0] * 10 * sizeof(float));
    _Float16* c2Buf   = (_Float16*)suballoc((size_t)B * C2H[0] * C2W[0] * 16 * sizeof(_Float16));
    _Float16* w3pack  = (_Float16*)suballoc(5 * 2 * 32 * 16 * sizeof(_Float16));
    // c1Buf doubles as the H-resize intermediate (B*648*1080*3 floats < c1 size)

    pack_w3_kernel<<<20, 256, 0, stream>>>(w3, w3pack);

    float* regBase  = out + (size_t)B * totalAll * 5;
    float* maskBase = out + (size_t)B * totalAll * 9;

    for (int s = 0; s < ns; ++s) {
        int hs = HS[s], ws = WS[s];
        float* tmp = c1Buf;  // reused before conv1 overwrites it

        long long n1 = (long long)B * hs * W;
        resize_h_kernel<<<cdivu(n1, 256), 256, 0, stream>>>(inp, tmp, hs, H, W, B);

        long long n2 = (long long)B * hs * ws;
        resize_w_kernel<<<cdivu(n2, 256), 256, 0, stream>>>(tmp, imgBuf, hs, W, ws, B);

        long long n3 = (long long)B * C1H[s] * C1W[s];
        conv1_kernel<<<cdivu(n3, 256), 256, 0, stream>>>(imgBuf, w1, b1, c1Buf,
                                                         hs, ws, C1H[s], C1W[s], B);

        long long n4 = (long long)B * PH[s] * PW[s];
        pool_kernel<<<cdivu(n4, 256), 256, 0, stream>>>(c1Buf, poolBuf,
                                                        C1H[s], C1W[s], PH[s], PW[s], B);

        long long n5 = (long long)B * C2H[s] * C2W[s];
        conv2_kernel<<<cdivu(n5, 256), 256, 0, stream>>>(poolBuf, w2, b2, c2Buf,
                                                         PH[s], PW[s], C2H[s], C2W[s], B);

        int tilesPerRow = (C3W[s] + 15) / 16;
        int nTiles = B * C3H[s] * tilesPerRow;
        unsigned blocks = cdivu(nTiles, 4);
        conv3_heads_kernel<<<blocks, 128, 0, stream>>>(
            c2Buf, w3pack, b3, wp, bp, wr, br,
            out, regBase, maskBase,
            C2H[s], C2W[s], C3H[s], C3W[s], tilesPerRow, nTiles,
            (float)scales[s], OFF[s], totalAll);
    }
}
